// EncodeProcessDecode_36593121362318
// MI455X (gfx1250) — matrix operand order repaired and checked by reading, not compiled, not run on hardware
//
#include <hip/hip_runtime.h>

// ---------------------------------------------------------------------------
// Types for gfx1250 WMMA (wave32)
// ---------------------------------------------------------------------------
typedef __bf16 bf16;
typedef __attribute__((ext_vector_type(16))) __bf16        v16bf;
typedef __attribute__((ext_vector_type(8)))  float         v8f;
typedef __attribute__((ext_vector_type(4)))  unsigned int  u32x4;

union Frag16 { v16bf v; u32x4 q[2]; };

// A-matrix 16x32 bf16 fragment (ISA 7.12.2): lane L<16 holds row L,
// elements 0..7 = K k0..k0+7, elements 8..15 = K k0+16..k0+23 (k0 = lane-half*8).
__device__ __forceinline__ v16bf load_a_frag(const bf16* __restrict__ row, int k0) {
  Frag16 f;
  f.q[0] = *reinterpret_cast<const u32x4*>(row + k0);
  f.q[1] = *reinterpret_cast<const u32x4*>(row + k0 + 16);
  return f.v;
}

// B-matrix 32x16 bf16 fragment: lane L<16 holds col L, K=0..7,16..23; lanes 16..31
// hold K=8..15,24..31 (same as A). p points at Wt[col*K + kchunk + kb].
__device__ __forceinline__ v16bf load_b_frag(const bf16* __restrict__ p) {
  Frag16 f;
  f.q[0] = *reinterpret_cast<const u32x4*>(p);
  f.q[1] = *reinterpret_cast<const u32x4*>(p + 16);
  return f.v;
}

// ---------------------------------------------------------------------------
// Layer-1 GEMM: Y = relu(X @ W1 + b1), X assembled from up to 3 gather
// segments of width 128 (bf16). Output bf16 [M,128].
// Block = 128 threads = 4 waves; each wave computes 32 rows x 128 cols
// (2 row sub-tiles) so each B fragment is reused by two WMMAs.
// ---------------------------------------------------------------------------
template <int NSEG>
__global__ __launch_bounds__(128) void mlp_l1_wmma(
    const bf16* __restrict__ s0, const long long* __restrict__ i0,
    const bf16* __restrict__ s1, const long long* __restrict__ i1,
    const bf16* __restrict__ s2, const long long* __restrict__ i2,
    const bf16* __restrict__ Wt,   // [128][K] bf16 (transposed weights)
    const float* __restrict__ b1,  // [128]
    bf16* __restrict__ out,        // [M][128]
    int M) {
  constexpr int K = NSEG * 128;
  const int lane = threadIdx.x & 31;
  const int wave = threadIdx.x >> 5;
  const int rowbase = (blockIdx.x * 4 + wave) * 32;
  if (rowbase >= M) return;                 // wave-uniform: EXEC stays all-ones
  const int half = lane >> 4;
  const int n0   = lane & 15;
  const int ka = half * 8;                  // A fragment K offset per lane half
  const int kb = half * 8;                  // B fragment K offset per lane half

  const bf16*      srcs[3] = { s0, s1, s2 };
  const long long* idxs[3] = { i0, i1, i2 };
  const bf16* arow[2][NSEG];
#pragma unroll
  for (int r = 0; r < 2; ++r) {
    const int mrow = rowbase + r * 16 + n0;
    const int mclamp = mrow < M ? mrow : (M - 1);
#pragma unroll
    for (int s = 0; s < NSEG; ++s) {
      long long ri = idxs[s] ? idxs[s][mclamp] : (long long)mclamp;
      arow[r][s] = srcs[s] + ri * 128;
    }
  }

  // C initialized with the bias: every element of acc[r][t] is column t*16+n0.
  v8f acc[2][8];
#pragma unroll
  for (int t = 0; t < 8; ++t) {
    const float bias = b1[t * 16 + n0];
#pragma unroll
    for (int r = 0; r < 2; ++r)
#pragma unroll
      for (int e = 0; e < 8; ++e) acc[r][t][e] = bias;
  }

#pragma unroll
  for (int s = 0; s < NSEG; ++s) {
#pragma unroll
    for (int kc = 0; kc < 128; kc += 32) {
      v16bf a0 = load_a_frag(arow[0][s], kc + ka);
      v16bf a1 = load_a_frag(arow[1][s], kc + ka);
#pragma unroll
      for (int t = 0; t < 8; ++t) {
        const bf16* bp = Wt + (size_t)(t * 16 + n0) * K + s * 128 + kc + kb;
        v16bf b = load_b_frag(bp);
        acc[0][t] = __builtin_amdgcn_wmma_f32_16x16x32_bf16(
            false, a0, false, b, (short)0, acc[0][t], false, false);
        acc[1][t] = __builtin_amdgcn_wmma_f32_16x16x32_bf16(
            false, a1, false, b, (short)0, acc[1][t], false, false);
      }
    }
  }

  // Epilogue: ReLU, store bf16. C/D layout: VGPR e -> row base+half*8+e,
  // col = t*16 + (lane&15).
#pragma unroll
  for (int r = 0; r < 2; ++r) {
#pragma unroll
    for (int t = 0; t < 8; ++t) {
      const int col = t * 16 + n0;
#pragma unroll
      for (int e = 0; e < 8; ++e) {
        const int row = rowbase + r * 16 + half * 8 + e;
        if (row < M) {
          float y = acc[r][t][e];
          y = y > 0.0f ? y : 0.0f;
          out[(size_t)row * 128 + col] = (bf16)y;
        }
      }
    }
  }
}

// ---------------------------------------------------------------------------
// Layer-2 GEMM: Y = LN(H @ W2 + b2), K = 128. Epilogue variants.
// ---------------------------------------------------------------------------
constexpr int EPI_LN_WRITE   = 0;  // write f32 + bf16
constexpr int EPI_LN_RES     = 1;  // dst_f32 += Y (residual master), bf16 shadow; opt aux = bf16(Y)
constexpr int EPI_LN_SCATTER = 2;  // atomicAdd into dst_f32[scat_idx[row]*128 + col]

template <int EPI>
__global__ __launch_bounds__(128) void mlp_l2_wmma(
    const bf16* __restrict__ hin,   // [M][128] bf16
    const bf16* __restrict__ Wt,    // [128][128] bf16 transposed
    const float* __restrict__ b2,
    const float* __restrict__ gamma,
    const float* __restrict__ beta,
    float* __restrict__ dst_f32,
    bf16*  __restrict__ dst_bf16,
    bf16*  __restrict__ aux_bf16,
    const long long* __restrict__ scat_idx,
    int M) {
  const int lane = threadIdx.x & 31;
  const int wave = threadIdx.x >> 5;
  const int rowbase = (blockIdx.x * 4 + wave) * 32;
  if (rowbase >= M) return;
  const int half = lane >> 4;
  const int n0   = lane & 15;
  const int ka = half * 8;
  const int kb = half * 8;

  const bf16* arow[2];
#pragma unroll
  for (int r = 0; r < 2; ++r) {
    const int mrow = rowbase + r * 16 + n0;
    const int mclamp = mrow < M ? mrow : (M - 1);
    arow[r] = hin + (size_t)mclamp * 128;
  }

  float g[8], bt[8];
  v8f acc[2][8];
#pragma unroll
  for (int t = 0; t < 8; ++t) {
    const int col = t * 16 + n0;
    const float bias = b2[col];
    g[t]  = gamma[col];
    bt[t] = beta[col];
#pragma unroll
    for (int r = 0; r < 2; ++r)
#pragma unroll
      for (int e = 0; e < 8; ++e) acc[r][t][e] = bias;
  }

#pragma unroll
  for (int kc = 0; kc < 128; kc += 32) {
    v16bf a0 = load_a_frag(arow[0], kc + ka);
    v16bf a1 = load_a_frag(arow[1], kc + ka);
#pragma unroll
    for (int t = 0; t < 8; ++t) {
      const bf16* bp = Wt + (size_t)(t * 16 + n0) * 128 + kc + kb;
      v16bf b = load_b_frag(bp);
      acc[0][t] = __builtin_amdgcn_wmma_f32_16x16x32_bf16(
          false, a0, false, b, (short)0, acc[0][t], false, false);
      acc[1][t] = __builtin_amdgcn_wmma_f32_16x16x32_bf16(
          false, a1, false, b, (short)0, acc[1][t], false, false);
    }
  }

  // Row-wise LayerNorm: each row lives in one lane-half (16 lanes x 8 tiles).
#pragma unroll
  for (int r = 0; r < 2; ++r) {
#pragma unroll
    for (int e = 0; e < 8; ++e) {
      float s1 = 0.0f, s2 = 0.0f;
#pragma unroll
      for (int t = 0; t < 8; ++t) { float v = acc[r][t][e]; s1 += v; s2 += v * v; }
#pragma unroll
      for (int m = 1; m < 16; m <<= 1) {   // butterfly within each 16-lane half
        s1 += __shfl_xor(s1, m, 32);
        s2 += __shfl_xor(s2, m, 32);
      }
      const float mean = s1 * (1.0f / 128.0f);
      const float var  = s2 * (1.0f / 128.0f) - mean * mean;
      const float rstd = rsqrtf(var + 1e-5f);
      const int row = rowbase + r * 16 + half * 8 + e;
      if (row >= M) continue;

      if (EPI == EPI_LN_SCATTER) {
        const long long c = scat_idx[row];
        float* aggrow = dst_f32 + (size_t)c * 128;
#pragma unroll
        for (int t = 0; t < 8; ++t) {
          const float y = (acc[r][t][e] - mean) * rstd * g[t] + bt[t];
          atomicAdd(aggrow + t * 16 + n0, y);
        }
      } else {
#pragma unroll
        for (int t = 0; t < 8; ++t) {
          const float y = (acc[r][t][e] - mean) * rstd * g[t] + bt[t];
          const size_t off = (size_t)row * 128 + t * 16 + n0;
          if (EPI == EPI_LN_WRITE) {
            dst_f32[off]  = y;
            dst_bf16[off] = (bf16)y;
          } else {  // EPI_LN_RES
            if (aux_bf16) aux_bf16[off] = (bf16)y;
            const float ny = y + dst_f32[off];
            dst_f32[off]  = ny;
            dst_bf16[off] = (bf16)ny;
          }
        }
      }
    }
  }
}

// ---------------------------------------------------------------------------
// Small-K / utility kernels (VALU; <0.3% of total FLOPs)
// ---------------------------------------------------------------------------
__global__ void node_enc_l1(const float* __restrict__ x, const float* __restrict__ phi,
                            const float* __restrict__ w1, const float* __restrict__ b1,
                            bf16* __restrict__ out, int N) {
  int tid = blockIdx.x * blockDim.x + threadIdx.x;
  if (tid >= N * 128) return;
  int n = tid >> 7, h = tid & 127;
  const float* xr = x + (size_t)n * 12;
  float s = b1[h];
#pragma unroll
  for (int k = 0; k < 12; ++k) s += xr[k] * w1[k * 128 + h];
  float load = (xr[7] == 1.0f) ? phi[0] : 0.0f;   // swelling load input column
  s += load * w1[12 * 128 + h];
  s = s > 0.0f ? s : 0.0f;
  out[tid] = (bf16)s;
}

__global__ void edge_enc_l1(const float* __restrict__ ea,
                            const float* __restrict__ w1, const float* __restrict__ b1,
                            bf16* __restrict__ out, int E) {
  int tid = blockIdx.x * blockDim.x + threadIdx.x;
  if (tid >= E * 128) return;
  int e = tid >> 7, h = tid & 127;
  const float* er = ea + (size_t)e * 3;
  float s = b1[h];
#pragma unroll
  for (int k = 0; k < 3; ++k) s += er[k] * w1[k * 128 + h];
  s = s > 0.0f ? s : 0.0f;
  out[tid] = (bf16)s;
}

__global__ void dec_l2(const bf16* __restrict__ h, const float* __restrict__ w2,
                       const float* __restrict__ b2, float* __restrict__ out, int N) {
  int tid = blockIdx.x * blockDim.x + threadIdx.x;
  if (tid >= N * 3) return;
  int n = tid / 3, o = tid - n * 3;
  const bf16* hr = h + (size_t)n * 128;
  float s = b2[o];
#pragma unroll 8
  for (int k = 0; k < 128; ++k) s += (float)hr[k] * w2[k * 3 + o];
  out[tid] = s;
}

__global__ void zero_f32(float* __restrict__ p, int n) {
  int tid = blockIdx.x * blockDim.x + threadIdx.x;
  if (tid < n) p[tid] = 0.0f;
}

__global__ void cvt_f32_bf16(const float* __restrict__ in, bf16* __restrict__ out, int n) {
  int tid = blockIdx.x * blockDim.x + threadIdx.x;
  if (tid < n) out[tid] = (bf16)in[tid];
}

// W [K][128] f32 (row-major, fan_in x fan_out)  ->  Wt [128][K] bf16 (transposed)
__global__ void wt_cvt(const float* __restrict__ W, bf16* __restrict__ Wt, int K) {
  int tid = blockIdx.x * blockDim.x + threadIdx.x;
  if (tid >= K * 128) return;
  int k = tid >> 7, n = tid & 127;
  Wt[(size_t)n * K + k] = (bf16)W[(size_t)k * 128 + n];
}

// ---------------------------------------------------------------------------
// Host launch
// ---------------------------------------------------------------------------
#define CEILDIV(a, b) (((a) + (b) - 1) / (b))

extern "C" void kernel_launch(void* const* d_in, const int* in_sizes, int n_in,
                              void* d_out, int out_size, void* d_ws, size_t ws_size,
                              hipStream_t stream) {
  (void)in_sizes; (void)out_size;
  constexpr int N = 25000, E = 150000, H = 128;
  if (n_in < 74) return;

  // ---- inputs (setup_inputs order) ----
  const float*     x     = (const float*)d_in[0];
  const long long* eidx  = (const long long*)d_in[1];   // int64 [2, E]
  const float*     eattr = (const float*)d_in[2];
  const float*     phi   = (const float*)d_in[3];
  const long long* row_i = eidx;         // edge_index[0]
  const long long* col_i = eidx + E;     // edge_index[1]

  // ---- params: JAX sorted-key pytree flatten ----
  // dec < edge_enc < node_enc < procs ; per MLP: [beta, gamma,] l1.b, l1.w, l2.b, l2.w
  const float* dec_l1b = (const float*)d_in[4];
  const float* dec_l1w = (const float*)d_in[5];   // [128,128]
  const float* dec_l2b = (const float*)d_in[6];
  const float* dec_l2w = (const float*)d_in[7];   // [128,3]
  const float* ee_beta = (const float*)d_in[8];
  const float* ee_gam  = (const float*)d_in[9];
  const float* ee_l1b  = (const float*)d_in[10];
  const float* ee_l1w  = (const float*)d_in[11];  // [3,128]
  const float* ee_l2b  = (const float*)d_in[12];
  const float* ee_l2w  = (const float*)d_in[13];  // [128,128]
  const float* ne_beta = (const float*)d_in[14];
  const float* ne_gam  = (const float*)d_in[15];
  const float* ne_l1b  = (const float*)d_in[16];
  const float* ne_l1w  = (const float*)d_in[17];  // [13,128]
  const float* ne_l2b  = (const float*)d_in[18];
  const float* ne_l2w  = (const float*)d_in[19];  // [128,128]
  // procs[s] : edge < msg < node, 6 leaves each
  const float* pr[3][3][6];
  for (int s = 0; s < 3; ++s)
    for (int m = 0; m < 3; ++m)
      for (int f = 0; f < 6; ++f)
        pr[s][m][f] = (const float*)d_in[20 + s * 18 + m * 6 + f];

  // ---- workspace carve ----
  size_t off = 0;
  auto carve = [&](size_t bytes) -> char* {
    char* p = (char*)d_ws + off;
    off += (bytes + 255) & ~(size_t)255;
    return p;
  };
  float* xh_f32  = (float*)carve((size_t)N * H * 4);
  bf16*  xh_bf   = (bf16*) carve((size_t)N * H * 2);
  float* eh_f32  = (float*)carve((size_t)E * H * 4);
  bf16*  eh_bf   = (bf16*) carve((size_t)E * H * 2);
  bf16*  eupd_bf = (bf16*) carve((size_t)E * H * 2);
  bf16*  hE      = (bf16*) carve((size_t)E * H * 2);
  bf16*  hN      = (bf16*) carve((size_t)N * H * 2);
  float* agg_f32 = (float*)carve((size_t)N * H * 4);
  bf16*  agg_bf  = (bf16*) carve((size_t)N * H * 2);
  bf16*  WtNE2   = (bf16*) carve((size_t)128 * 128 * 2);
  bf16*  WtEE2   = (bf16*) carve((size_t)128 * 128 * 2);
  bf16*  WtDEC1  = (bf16*) carve((size_t)128 * 128 * 2);
  bf16 *WtE1[3], *WtE2[3], *WtM1[3], *WtM2[3], *WtN1[3], *WtN2[3];
  for (int s = 0; s < 3; ++s) {
    WtE1[s] = (bf16*)carve((size_t)384 * 128 * 2);
    WtE2[s] = (bf16*)carve((size_t)128 * 128 * 2);
    WtM1[s] = (bf16*)carve((size_t)256 * 128 * 2);
    WtM2[s] = (bf16*)carve((size_t)128 * 128 * 2);
    WtN1[s] = (bf16*)carve((size_t)256 * 128 * 2);
    WtN2[s] = (bf16*)carve((size_t)128 * 128 * 2);
  }
  if (off > ws_size) return;  // workspace too small; bail deterministically

  // ---- weight convert + transpose (idempotent each call) ----
  auto cvtw = [&](const float* W, bf16* Wt, int K) {
    wt_cvt<<<CEILDIV(K * 128, 256), 256, 0, stream>>>(W, Wt, K);
  };
  cvtw(ne_l2w, WtNE2, 128);
  cvtw(ee_l2w, WtEE2, 128);
  cvtw(dec_l1w, WtDEC1, 128);
  for (int s = 0; s < 3; ++s) {
    cvtw(pr[s][0][3], WtE1[s], 384);
    cvtw(pr[s][0][5], WtE2[s], 128);
    cvtw(pr[s][1][3], WtM1[s], 256);
    cvtw(pr[s][1][5], WtM2[s], 128);
    cvtw(pr[s][2][3], WtN1[s], 256);
    cvtw(pr[s][2][5], WtN2[s], 128);
  }

  // 4 waves/block, 32 rows/wave -> 128 rows per block
  const int gN = CEILDIV(N, 128), gE = CEILDIV(E, 128);

  // ---- encoders ----
  node_enc_l1<<<CEILDIV(N * H, 256), 256, 0, stream>>>(x, phi, ne_l1w, ne_l1b, hN, N);
  mlp_l2_wmma<EPI_LN_WRITE><<<gN, 128, 0, stream>>>(
      hN, WtNE2, ne_l2b, ne_gam, ne_beta, xh_f32, xh_bf, nullptr, nullptr, N);
  edge_enc_l1<<<CEILDIV(E * H, 256), 256, 0, stream>>>(eattr, ee_l1w, ee_l1b, hE, E);
  mlp_l2_wmma<EPI_LN_WRITE><<<gE, 128, 0, stream>>>(
      hE, WtEE2, ee_l2b, ee_gam, ee_beta, eh_f32, eh_bf, nullptr, nullptr, E);

  // ---- process steps ----
  for (int s = 0; s < 3; ++s) {
    const float *eb = pr[s][0][0], *eg = pr[s][0][1], *e1b = pr[s][0][2], *e2b = pr[s][0][4];
    const float *mb = pr[s][1][0], *mg = pr[s][1][1], *m1b = pr[s][1][2], *m2b = pr[s][1][4];
    const float *nb = pr[s][2][0], *ng = pr[s][2][1], *n1b = pr[s][2][2], *n2b = pr[s][2][4];

    // edge MLP: cat(x_h[row], x_h[col], e_h) -> e_upd ; e_h += e_upd
    mlp_l1_wmma<3><<<gE, 128, 0, stream>>>(
        xh_bf, row_i, xh_bf, col_i, eh_bf, nullptr, WtE1[s], e1b, hE, E);
    mlp_l2_wmma<EPI_LN_RES><<<gE, 128, 0, stream>>>(
        hE, WtE2[s], e2b, eg, eb, eh_f32, eh_bf, eupd_bf, nullptr, E);

    // msg MLP: cat(x_h[row], e_upd) -> msg ; agg = segment_sum(msg, col)
    mlp_l1_wmma<2><<<gE, 128, 0, stream>>>(
        xh_bf, row_i, eupd_bf, nullptr, nullptr, nullptr, WtM1[s], m1b, hE, E);
    zero_f32<<<CEILDIV(N * H, 256), 256, 0, stream>>>(agg_f32, N * H);
    mlp_l2_wmma<EPI_LN_SCATTER><<<gE, 128, 0, stream>>>(
        hE, WtM2[s], m2b, mg, mb, agg_f32, nullptr, nullptr, col_i, E);
    cvt_f32_bf16<<<CEILDIV(N * H, 256), 256, 0, stream>>>(agg_f32, agg_bf, N * H);

    // node MLP: cat(x_h, agg) -> x_h += ...
    mlp_l1_wmma<2><<<gN, 128, 0, stream>>>(
        xh_bf, nullptr, agg_bf, nullptr, nullptr, nullptr, WtN1[s], n1b, hN, N);
    mlp_l2_wmma<EPI_LN_RES><<<gN, 128, 0, stream>>>(
        hN, WtN2[s], n2b, ng, nb, xh_f32, xh_bf, nullptr, nullptr, N);
  }

  // ---- decoder ----
  mlp_l1_wmma<1><<<gN, 128, 0, stream>>>(
      xh_bf, nullptr, nullptr, nullptr, nullptr, nullptr, WtDEC1, dec_l1b, hN, N);
  dec_l2<<<CEILDIV(N * 3, 256), 256, 0, stream>>>(hN, dec_l2w, dec_l2b, (float*)d_out, N);
}
